// HGT_52467320487957
// MI455X (gfx1250) — matrix operand order repaired and checked by reading, not compiled, not run on hardware
//
#include <hip/hip_runtime.h>
#include <math.h>

// ---------------------------------------------------------------------------
// HGT on gfx1250 (MI455X). Dense projections via v_wmma_f32_16x16x32_bf16 with
// pre-converted bf16 operands (A row-major, W transposed to [N][K]) so the
// GEMM inner loop is pure b128 loads + WMMA. Edge softmax/scatter on VALU +
// f32 atomics. Compile-only target.
// ---------------------------------------------------------------------------

typedef __attribute__((ext_vector_type(16))) __bf16 v16bf;
typedef __attribute__((ext_vector_type(8)))  __bf16 v8bf;
typedef __attribute__((ext_vector_type(8)))  float  v8f;

__device__ __forceinline__ __bf16 to_bf16(float f) {
    // round-to-nearest-even f32 -> bf16 (bit-level)
    unsigned u = __float_as_uint(f);
    unsigned r = (u + 0x7fffu + ((u >> 16) & 1u)) >> 16;
    unsigned short hs = (unsigned short)r;
    __bf16 h;
    __builtin_memcpy(&h, &hs, 2);
    return h;
}

__device__ __forceinline__ float gelu_f(float x) {
    return 0.5f * x * (1.0f + erff(x * 0.7071067811865476f));
}

__device__ __forceinline__ void atomicMaxF(float* addr, float v) {
    // monotonic float max via int atomics (addr init to -inf, no NaNs)
    if (v >= 0.0f) atomicMax((int*)addr, __float_as_int(v));
    else           atomicMin((unsigned int*)addr, __float_as_uint(v));
}

// A fragment (16x32 bf16): M = lane&15; lanes<16 hold K in kk..kk+15,
// lanes>=16 hold K in kk+16..kk+31.  Row stride 128 bf16.
// Two aligned 16-byte loads per lane.
__device__ __forceinline__ v16bf load_a_bf(const __bf16* __restrict__ A, int row,
                                           int kk, int lane) {
    int base = (lane & 16) ? 16 : 0;
    const __bf16* p = A + (size_t)row * 128 + kk + base;
    union { v16bf v; struct { v8bf lo; v8bf hi; } s; } u;
    u.s.lo = *(const v8bf*)(p);
    u.s.hi = *(const v8bf*)(p + 8);
    return u.v;
}

// B fragment (32x16 bf16) from W^T laid out [N][K]: col = per-lane, K contiguous.
// lanes<16 hold K=kk..kk+15, lanes>=16 hold K=kk+16..kk+31.  One 32-byte load.
__device__ __forceinline__ v16bf load_b_bf(const __bf16* __restrict__ Wt, int col,
                                           int kk, int ldk, int lane) {
    int kb = (lane & 16) ? 16 : 0;
    return *(const v16bf*)(Wt + (size_t)col * ldk + kk + kb);
}

// C[M x 128] = A[M x 128](bf16) @ W(bf16, stored transposed [128][128]) + bias.
// One wave per 16x16 tile; 8 waves/block cover N=128; 4 WMMAs (K=128) per tile.
// out_bf selects f32 or bf16 output.
__global__ __launch_bounds__(256) void k_gemm128_bf(const __bf16* __restrict__ A,
                                                    const __bf16* __restrict__ Wt,
                                                    const float* __restrict__ bias,
                                                    float* __restrict__ Cf,
                                                    __bf16* __restrict__ Cb,
                                                    int M, int out_bf) {
    int lane = threadIdx.x & 31;
    int wave = threadIdx.x >> 5;
    int row0 = blockIdx.x * 16;
    int n0   = wave * 16;
    int rowA = row0 + (lane & 15);
    if (rowA >= M) rowA = M - 1;
    int col = n0 + (lane & 15);

    v8f acc = {};
#pragma unroll
    for (int kk = 0; kk < 128; kk += 32) {
        v16bf a = load_a_bf(A, rowA, kk, lane);
        v16bf b = load_b_bf(Wt, col, kk, 128, lane);
        acc = __builtin_amdgcn_wmma_f32_16x16x32_bf16(false, a, false, b,
                                                      (short)0, acc, false, false);
    }
    int rb = (lane & 16) ? 8 : 0;
    float bv = bias[col];
#pragma unroll
    for (int r = 0; r < 8; ++r) {
        int rr = row0 + r + rb;
        if (rr < M) {
            float v = acc[r] + bv;
            if (out_bf) Cb[(size_t)rr * 128 + col] = to_bf16(v);
            else        Cf[(size_t)rr * 128 + col] = v;
        }
    }
}

// Per-head relation transform: O[n, h*32+e] = sum_d X[n, h*32+d] * REL[h][d][e].
// X bf16 [M][128]; RELT bf16 pre-transposed [4][32e][32d]. Wave w: h = w>>1,
// col-half = w&1. One WMMA (K=32) per wave; f32 output.
__global__ __launch_bounds__(256) void k_rel32_bf(const __bf16* __restrict__ X,
                                                  const __bf16* __restrict__ RELT,
                                                  float* __restrict__ O, int M) {
    int lane = threadIdx.x & 31;
    int wave = threadIdx.x >> 5;
    int h    = wave >> 1;
    int n0   = (wave & 1) * 16;
    int row0 = blockIdx.x * 16;
    int rowA = row0 + (lane & 15);
    if (rowA >= M) rowA = M - 1;
    int colL = n0 + (lane & 15);

    v16bf a = load_a_bf(X + h * 32, rowA, 0, lane);
    v16bf b = load_b_bf(RELT + h * 1024, colL, 0, 32, lane);
    v8f acc = {};
    acc = __builtin_amdgcn_wmma_f32_16x16x32_bf16(false, a, false, b,
                                                  (short)0, acc, false, false);
    int rb = (lane & 16) ? 8 : 0;
    int cc = h * 32 + colL;
#pragma unroll
    for (int r = 0; r < 8; ++r) {
        int rr = row0 + r + rb;
        if (rr < M) O[(size_t)rr * 128 + cc] = acc[r];
    }
}

// elementwise f32 -> bf16 (optional exact GELU), done once per tensor
__global__ void k_cvt_bf16(const float* __restrict__ in, __bf16* __restrict__ out,
                           size_t n, int act) {
    size_t i = (size_t)blockIdx.x * blockDim.x + threadIdx.x;
    if (i >= n) return;
    float v = in[i];
    if (act) v = gelu_f(v);
    out[i] = to_bf16(v);
}

// batched K x N -> N x K transpose with bf16 conversion (weights; tiny)
__global__ void k_transpose_bf16(const float* __restrict__ in, __bf16* __restrict__ out,
                                 int K, int N, int batch) {
    int total = K * N * batch;
    int idx = blockIdx.x * blockDim.x + threadIdx.x;
    if (idx >= total) return;
    int b = idx / (K * N);
    int r = idx - b * (K * N);
    int k = r / N;
    int n = r - k * N;
    out[(size_t)b * K * N + (size_t)n * K + k] = to_bf16(in[idx]);
}

__global__ void k_fill(float* __restrict__ p, float v, size_t n) {
    size_t i = (size_t)blockIdx.x * blockDim.x + threadIdx.x;
    if (i < n) p[i] = v;
}

// Pass 1: logits + per-(dst,head) running max
__global__ void k_edge_logit_max(const int* __restrict__ e, int E,
                                 const float* __restrict__ Q,
                                 const float* __restrict__ KT,
                                 const float* __restrict__ prel,
                                 float* __restrict__ LG, float* __restrict__ MX) {
    int idx = blockIdx.x * blockDim.x + threadIdx.x;
    if (idx >= E * 4) return;
    int ei = idx >> 2, h = idx & 3;
    int src = e[ei], dst = e[E + ei];
    const float* q = Q + (size_t)dst * 128 + h * 32;
    const float* k = KT + (size_t)src * 128 + h * 32;
    float acc = 0.0f;
#pragma unroll
    for (int d = 0; d < 32; ++d) acc = fmaf(q[d], k[d], acc);
    float logit = acc * prel[h] * 0.17677669529663687f;  // 1/sqrt(32)
    LG[idx] = logit;
    atomicMaxF(&MX[dst * 4 + h], logit);
}

// Pass 2: exp(logit - max), per-(dst,head) sum
__global__ void k_edge_expsum(const int* __restrict__ e, int E,
                              float* __restrict__ LG,
                              const float* __restrict__ MX,
                              float* __restrict__ SX) {
    int idx = blockIdx.x * blockDim.x + threadIdx.x;
    if (idx >= E * 4) return;
    int ei = idx >> 2, h = idx & 3;
    int dst = e[E + ei];
    float ev = __expf(LG[idx] - MX[dst * 4 + h]);
    LG[idx] = ev;
    atomicAdd(&SX[dst * 4 + h], ev);
}

// Pass 3: normalize + weighted scatter-add of vt messages
__global__ void k_edge_scatter(const int* __restrict__ e, int E,
                               const float* __restrict__ VT,
                               const float* __restrict__ LG,
                               const float* __restrict__ SX,
                               float* __restrict__ AC) {
    int idx = blockIdx.x * blockDim.x + threadIdx.x;
    if (idx >= E * 4) return;
    int ei = idx >> 2, h = idx & 3;
    int src = e[ei], dst = e[E + ei];
    float w = LG[idx] / SX[dst * 4 + h];
    const float* v = VT + (size_t)src * 128 + h * 32;
    float* a = AC + (size_t)dst * 128 + h * 32;
#pragma unroll
    for (int d = 0; d < 32; ++d) atomicAdd(&a[d], v[d] * w);
}

// out = g*o + (1-g)*x, g = sigmoid(skip[L][type]), type by row threshold
__global__ void k_mix(const float* __restrict__ o, const float* __restrict__ x,
                      float* __restrict__ out, const float* __restrict__ skipL,
                      int NP, size_t n) {
    size_t i = (size_t)blockIdx.x * blockDim.x + threadIdx.x;
    if (i >= n) return;
    int row = (int)(i >> 7);
    int t = (row < NP) ? 0 : 1;
    float g = 1.0f / (1.0f + __expf(-skipL[t]));
    out[i] = g * o[i] + (1.0f - g) * x[i];
}

extern "C" void kernel_launch(void* const* d_in, const int* in_sizes, int n_in,
                              void* d_out, int out_size, void* d_ws, size_t ws_size,
                              hipStream_t stream) {
    const float* x_paper  = (const float*)d_in[0];
    const float* x_author = (const float*)d_in[1];
    const int* e_ptr[3]   = {(const int*)d_in[2], (const int*)d_in[3], (const int*)d_in[4]};
    const float* lin_W = (const float*)d_in[5];
    const float* lin_b = (const float*)d_in[6];
    const float* k_W   = (const float*)d_in[7];
    const float* k_b   = (const float*)d_in[8];
    const float* q_W   = (const float*)d_in[9];
    const float* q_b   = (const float*)d_in[10];
    const float* v_W   = (const float*)d_in[11];
    const float* v_b   = (const float*)d_in[12];
    const float* a_W   = (const float*)d_in[13];
    const float* a_b   = (const float*)d_in[14];
    const float* skip  = (const float*)d_in[15];
    const float* a_rel = (const float*)d_in[16];
    const float* m_rel = (const float*)d_in[17];
    const float* p_rel = (const float*)d_in[18];

    const int NP = in_sizes[0] / 128;      // 100000
    const int NA = in_sizes[1] / 128;      // 50000
    const int NT = NP + NA;                // 150000
    const int E  = in_sizes[2] / 2;        // 200000

    // ---- workspace layout: f32 region first, then bf16 region ----
    float* wsf = (float*)d_ws;
    size_t off = 0;
    float* XS = wsf + off; off += (size_t)NT * 128;   // current features (f32)
    float* Qb = wsf + off; off += (size_t)NT * 128;   // Q (f32; aliased as a_lin output)
    float* AC = wsf + off; off += (size_t)NT * 128;   // message accumulator (f32 atomics)
    float* KT = wsf + off; off += (size_t)NP * 128;   // relation-transformed K (f32)
    float* VT = wsf + off; off += (size_t)NP * 128;   // relation-transformed V (f32)
    float* LG = wsf + off; off += (size_t)E * 4;      // logits -> exp weights
    float* MX = wsf + off; off += (size_t)NP * 4;     // segment max
    float* SX = wsf + off; off += (size_t)NP * 4;     // segment sum
    float* Ob = Qb;                                   // a_lin output aliases Q

    __bf16* wsb = (__bf16*)(wsf + off);
    size_t ob = 0;
    __bf16* XSb = wsb + ob; ob += (size_t)NT * 128;   // bf16 copy of XS (GEMM A)
    __bf16* Kb  = wsb + ob; ob += (size_t)NT * 128;   // K (bf16, feeds rel WMMA only)
    __bf16* Vb  = wsb + ob; ob += (size_t)NT * 128;   // V (bf16, feeds rel WMMA only)
    __bf16* ACb = wsb + ob; ob += (size_t)NT * 128;   // gelu(AC) bf16 (a_lin A)
    __bf16* WT_lin = wsb + ob; ob += (size_t)2 * 16384;   // transposed bf16 weights
    __bf16* WT_k   = wsb + ob; ob += (size_t)4 * 16384;
    __bf16* WT_q   = wsb + ob; ob += (size_t)4 * 16384;
    __bf16* WT_v   = wsb + ob; ob += (size_t)4 * 16384;
    __bf16* WT_a   = wsb + ob; ob += (size_t)4 * 16384;
    __bf16* RT_a   = wsb + ob; ob += (size_t)24 * 1024;   // [L][et][h][32e][32d]
    __bf16* RT_m   = wsb + ob; ob += (size_t)24 * 1024;
    (void)ws_size; (void)n_in; (void)out_size;

    const dim3 blk(256);
    auto gemm_f = [&](const __bf16* A, const __bf16* Wt, const float* b, float* C, int M) {
        k_gemm128_bf<<<dim3((M + 15) / 16), blk, 0, stream>>>(A, Wt, b, C, nullptr, M, 0);
    };
    auto gemm_b = [&](const __bf16* A, const __bf16* Wt, const float* b, __bf16* C, int M) {
        k_gemm128_bf<<<dim3((M + 15) / 16), blk, 0, stream>>>(A, Wt, b, nullptr, C, M, 1);
    };
    auto cvt = [&](const float* in, __bf16* out, size_t n, int act) {
        k_cvt_bf16<<<dim3((unsigned)((n + 255) / 256)), blk, 0, stream>>>(in, out, n, act);
    };
    auto fill = [&](float* p, float v, size_t n) {
        k_fill<<<dim3((unsigned)((n + 255) / 256)), blk, 0, stream>>>(p, v, n);
    };
    auto transpose = [&](const float* in, __bf16* out, int K, int N, int batch) {
        int total = K * N * batch;
        k_transpose_bf16<<<dim3((total + 255) / 256), blk, 0, stream>>>(in, out, K, N, batch);
    };

    // ---- pre-transpose all weights to bf16 [N][K] (tiny, L2-resident) ----
    transpose(lin_W, WT_lin, 128, 128, 2);
    transpose(k_W,   WT_k,   128, 128, 4);
    transpose(q_W,   WT_q,   128, 128, 4);
    transpose(v_W,   WT_v,   128, 128, 4);
    transpose(a_W,   WT_a,   128, 128, 4);
    transpose(a_rel, RT_a,   32,  32,  24);
    transpose(m_rel, RT_m,   32,  32,  24);

    // ---- input projection ----
    cvt(x_paper,  XSb,                    (size_t)NP * 128, 0);
    cvt(x_author, XSb + (size_t)NP * 128, (size_t)NA * 128, 0);
    gemm_f(XSb,                    WT_lin,         lin_b,       XS,                    NP);
    gemm_f(XSb + (size_t)NP * 128, WT_lin + 16384, lin_b + 128, XS + (size_t)NP * 128, NA);
    cvt(XS, XSb, (size_t)NT * 128, 0);

    const int spec_s[3] = {0, 1, 0};   // EDGE_SPECS src types
    const int spec_d[3] = {0, 0, 1};   // EDGE_SPECS dst types

    for (int L = 0; L < 2; ++L) {
        // K/Q/V projections per node type (K,V straight to bf16; Q stays f32)
        for (int t = 0; t < 2; ++t) {
            size_t base = t ? (size_t)NP * 128 : 0;
            int M = t ? NA : NP;
            size_t wi = (size_t)(L * 2 + t) * 16384;
            size_t bi = (size_t)(L * 2 + t) * 128;
            gemm_b(XSb + base, WT_k + wi, k_b + bi, Kb + base, M);
            gemm_f(XSb + base, WT_q + wi, q_b + bi, Qb + base, M);
            gemm_b(XSb + base, WT_v + wi, v_b + bi, Vb + base, M);
        }
        fill(AC, 0.0f, (size_t)NT * 128);

        for (int et = 0; et < 3; ++et) {
            int st = spec_s[et], dt = spec_d[et];
            size_t sbase = st ? (size_t)NP * 128 : 0;
            size_t dbase = dt ? (size_t)NP * 128 : 0;
            int Ms = st ? NA : NP;
            int Md = dt ? NA : NP;
            const __bf16* rA = RT_a + (size_t)(L * 3 + et) * 4096;
            const __bf16* rM = RT_m + (size_t)(L * 3 + et) * 4096;
            const float*  pr = p_rel + (size_t)(L * 3 + et) * 4;

            k_rel32_bf<<<dim3((Ms + 15) / 16), blk, 0, stream>>>(Kb + sbase, rA, KT, Ms);
            k_rel32_bf<<<dim3((Ms + 15) / 16), blk, 0, stream>>>(Vb + sbase, rM, VT, Ms);

            fill(MX, -INFINITY, (size_t)Md * 4);
            fill(SX, 0.0f,      (size_t)Md * 4);

            dim3 eg((E * 4 + 255) / 256);
            k_edge_logit_max<<<eg, blk, 0, stream>>>(e_ptr[et], E, Qb + dbase, KT, pr, LG, MX);
            k_edge_expsum  <<<eg, blk, 0, stream>>>(e_ptr[et], E, LG, MX, SX);
            k_edge_scatter <<<eg, blk, 0, stream>>>(e_ptr[et], E, VT, LG, SX, AC + dbase);
        }

        // output projection: gelu(AC) -> bf16 once, then WMMA GEMM; skip-gate mix
        cvt(AC, ACb, (size_t)NT * 128, 1 /*gelu*/);
        for (int t = 0; t < 2; ++t) {
            size_t base = t ? (size_t)NP * 128 : 0;
            int M = t ? NA : NP;
            gemm_f(ACb + base, WT_a + (size_t)(L * 2 + t) * 16384,
                   a_b + (size_t)(L * 2 + t) * 128, Ob + base, M);
        }
        float* dst = (L == 1) ? (float*)d_out : XS;
        size_t n = (size_t)NT * 128;
        k_mix<<<dim3((unsigned)((n + 255) / 256)), blk, 0, stream>>>(Ob, XS, dst,
                                                                    skip + L * 2, NP, n);
        if (L == 0) cvt(XS, XSb, n, 0);  // refresh bf16 activations for next layer
    }
}
